// CausalSelfAttention_45930380264307
// MI455X (gfx1250) — matrix operand order repaired and checked by reading, not compiled, not run on hardware
//
#include <hip/hip_runtime.h>
#include <stdint.h>

// ---------------- problem constants (match reference) ----------------
#define BB    4
#define TT    2048
#define CCH   1024          // C (channels)
#define HHD   64            // head dim
#define NHEAD 16
#define MROWS (BB*TT)       // 8192
#define NQKV  (3*CCH)       // 3072

typedef __attribute__((ext_vector_type(16))) __bf16 v16bf;
typedef __attribute__((ext_vector_type(8)))  float  v8f;
typedef __attribute__((ext_vector_type(4)))  unsigned int u32x4;
typedef __attribute__((ext_vector_type(8)))  unsigned int u32x8;

static __device__ __forceinline__ unsigned short f2bf(float f) {
  unsigned u = __float_as_uint(f);
  unsigned r = u + 0x7FFFu + ((u >> 16) & 1u);   // round-to-nearest-even
  return (unsigned short)(r >> 16);
}

union AB16 { v16bf v; uint4 q[2]; unsigned short h[16]; };

// A-matrix fragment (16x32 bf16):
//   lanes 0-15 : row = lane,    K = 0..15
//   lanes 16-31: row = lane-16, K = 16..31
static __device__ __forceinline__ v16bf load_afrag(const unsigned short* base, int ld, int lane) {
  AB16 u;
  const unsigned short* p = base + (size_t)(lane & 15) * ld + (lane >> 4) * 16;
  u.q[0] = *reinterpret_cast<const uint4*>(p);        // K = half*16 .. +7
  u.q[1] = *reinterpret_cast<const uint4*>(p + 8);    // K = half*16+8 .. +7
  return u.v;
}

// B-matrix fragment (32x16 bf16): lane n -> column n; lanes 0-15 K=0..15,
// lanes 16-31 K=16..31, sequential within the lane. `base` points at
// element [col0][k0] of a row-major [N][K] array (pre-transposed weights).
static __device__ __forceinline__ v16bf load_bfrag(const unsigned short* base, int ld, int lane) {
  AB16 u;
  const unsigned short* p = base + (size_t)(lane & 15) * ld + (lane >> 4) * 16;
  u.q[0] = *reinterpret_cast<const uint4*>(p);
  u.q[1] = *reinterpret_cast<const uint4*>(p + 8);
  return u.v;
}

#define WMMA_BF16(a, b, c) \
  __builtin_amdgcn_wmma_f32_16x16x32_bf16(false, (a), false, (b), (short)0, (c), false, false)

// ---- Tensor Data Mover: issue a <=2D tile load described by D# groups 0/1 ----
// Group0 (ISA 8.3): [1:0]=count=1, [63:32]=lds_addr, [120:64]=global_addr, [127:126]=type=2.
// Group1 (ISA 8.4): [17:16]=data_size(1 -> 2B), [79:48]=tensor_dim0, [111:80]=tensor_dim1,
//                   [127:112]=tile_dim0, [143:128]=tile_dim1, [207:160]=tensor_dim0_stride.
static __device__ __forceinline__ void tdm_load_2d(unsigned lds_addr, unsigned long long gaddr,
                                                   u32x8 g1) {
  u32x4 g0;
  g0[0] = 1u;                                   // count=1, user descriptor
  g0[1] = lds_addr;                             // LDS byte address
  g0[2] = (unsigned)(gaddr & 0xffffffffull);    // global_addr[31:0]
  g0[3] = (unsigned)(gaddr >> 32) | 0x80000000u;// global_addr[56:32] | type=2
  asm volatile("tensor_load_to_lds %0, %1" : : "s"(g0), "s"(g1) : "memory");
}

// ---------------- conversion kernels ----------------
__global__ __launch_bounds__(256) void cvt_f32_bf16(const float* __restrict__ in,
                                                    unsigned short* __restrict__ out, int n) {
  int i = (blockIdx.x * 256 + threadIdx.x) * 4;
  if (i < n) {
    float4 f = *reinterpret_cast<const float4*>(in + i);
    union { unsigned short h[4]; uint2 u; } o;
    o.h[0] = f2bf(f.x); o.h[1] = f2bf(f.y); o.h[2] = f2bf(f.z); o.h[3] = f2bf(f.w);
    *reinterpret_cast<uint2*>(out + i) = o.u;
  }
}

// in: [K][N] fp32 row-major -> out: [N][K] bf16 row-major (transpose+convert)
__global__ __launch_bounds__(256) void cvt_transpose_bf16(const float* __restrict__ in,
                                                          unsigned short* __restrict__ out,
                                                          int K, int N) {
  int idx = blockIdx.x * 256 + threadIdx.x;   // over N*K, k fastest (coalesced writes)
  if (idx < N * K) {
    int nI = idx / K, kI = idx - nI * K;
    out[idx] = f2bf(in[(size_t)kI * N + nI]);
  }
}

// Shared D# group1 for a 64x32 bf16 tile out of an [8192][1024] bf16 matrix.
static __device__ __forceinline__ u32x8 g1_a_tile() {
  u32x8 g1;
  g1[0] = 0x00010000u;   // workgroup_mask=0, data_size=1 (2 bytes)
  g1[1] = 0x04000000u;   // tensor_dim0 = 1024 (bits 79:48, low half)
  g1[2] = 0x20000000u;   // tensor_dim0 hi=0 | tensor_dim1 = 8192 (low half)
  g1[3] = 0x00200000u;   // tensor_dim1 hi=0 | tile_dim0 = 32
  g1[4] = 64u;           // tile_dim1 = 64, tile_dim2 = 0
  g1[5] = 1024u;         // tensor_dim0_stride = 1024 elements
  g1[6] = 0u;
  g1[7] = 0u;
  return g1;
}

// ---------------- GEMM core: C[MxN] = A[MxK] * Wt[NxK]^T ----------------
// Block: 256 thr = 8 waves (2 x 4). Block tile 64(M) x 256(N), K-step 32.
// Wave tile 32x64: 2 A-frags x 4 B-frags = 8 WMMAs per K-step.
// A tiles (64x32 bf16, 4KB) are double-buffered and DMA'd by the Tensor Data
// Mover: tile i+1 is issued, then `s_wait_tensorcnt 1` guarantees tile i is
// resident (TDM completes in order) while tile i+1 overlaps the WMMAs.

// GEMM1: qkv = x @ w_attn + b_attn, scatter into q/k/v [B,H,T,HD] bf16
__global__ __launch_bounds__(256) void gemm_qkv_wmma(
    const unsigned short* __restrict__ A, const unsigned short* __restrict__ Wt,
    const float* __restrict__ bias,
    unsigned short* __restrict__ Qd, unsigned short* __restrict__ Kd,
    unsigned short* __restrict__ Vd) {
  const int K = CCH, KT = CCH / 32;
  int tid = threadIdx.x, lane = tid & 31, wave = tid >> 5;
  int waveM = wave & 1, waveN = wave >> 1;
  int m0 = blockIdx.x * 64;
  int n0 = blockIdx.y * 256 + waveN * 64;

  __shared__ unsigned short Atile[2][64 * 32];
  unsigned lds_a[2] = {(unsigned)(size_t)(void*)&Atile[0][0],
                       (unsigned)(size_t)(void*)&Atile[1][0]};
  u32x8 g1 = g1_a_tile();

  const v8f vz = {0.f,0.f,0.f,0.f,0.f,0.f,0.f,0.f};
  v8f acc[2][4] = {{vz, vz, vz, vz}, {vz, vz, vz, vz}};

  if (wave == 0)   // prologue: tile 0 in flight
    tdm_load_2d(lds_a[0], (unsigned long long)(size_t)&A[(size_t)m0 * K], g1);

  for (int kt = 0; kt < KT; kt++) {
    int k0 = kt * 32;
    if (wave == 0) {
      if (kt + 1 < KT) {   // issue next tile, then wait for current only
        tdm_load_2d(lds_a[(kt + 1) & 1],
                    (unsigned long long)(size_t)&A[(size_t)m0 * K + (k0 + 32)], g1);
        __builtin_amdgcn_s_wait_tensorcnt(1);
      } else {
        __builtin_amdgcn_s_wait_tensorcnt(0);
      }
    }
    if (kt + 1 < KT)   // prefetch next weight chunk (global_prefetch_b8)
      __builtin_prefetch(&Wt[(size_t)(n0 + (lane & 15)) * K + k0 + 32], 0, 1);
    __syncthreads();
    const unsigned short* at = &Atile[kt & 1][0];
    v16bf a0 = load_afrag(at + (waveM * 32) * 32, 32, lane);
    v16bf a1 = load_afrag(at + (waveM * 32 + 16) * 32, 32, lane);
#pragma unroll
    for (int j = 0; j < 4; j++) {
      v16bf b = load_bfrag(&Wt[(size_t)(n0 + j * 16) * K + k0], K, lane);
      acc[0][j] = WMMA_BF16(a0, b, acc[0][j]);
      acc[1][j] = WMMA_BF16(a1, b, acc[1][j]);
    }
    __syncthreads();   // protects buffer kt&1 before its next TDM write
  }

  // epilogue: bias add + scatter into [B,H,T,HD] bf16
  int n = lane & 15, half = lane >> 4;
#pragma unroll
  for (int j = 0; j < 4; j++) {
    int c = n0 + j * 16 + n;
    float bv = bias[c];
    int which = c >> 10, cc = c & 1023, h = cc >> 6, d = cc & 63;
    unsigned short* dst = (which == 0) ? Qd : (which == 1) ? Kd : Vd;
#pragma unroll
    for (int s = 0; s < 2; s++) {
#pragma unroll
      for (int r = 0; r < 8; r++) {
        int row = m0 + waveM * 32 + s * 16 + half * 8 + r;    // b*T + t
        int b = row >> 11, t = row & 2047;
        dst[(((size_t)(b * NHEAD + h)) * TT + t) * HHD + d] = f2bf(acc[s][j][r] + bv);
      }
    }
  }
}

// GEMM2: out = y @ w_proj + b_proj (fp32 out)
__global__ __launch_bounds__(256) void gemm_proj_wmma(
    const unsigned short* __restrict__ A, const unsigned short* __restrict__ Wt,
    const float* __restrict__ bias, float* __restrict__ out) {
  const int K = CCH, KT = CCH / 32;
  int tid = threadIdx.x, lane = tid & 31, wave = tid >> 5;
  int waveM = wave & 1, waveN = wave >> 1;
  int m0 = blockIdx.x * 64;
  int n0 = blockIdx.y * 256 + waveN * 64;

  __shared__ unsigned short Atile[2][64 * 32];
  unsigned lds_a[2] = {(unsigned)(size_t)(void*)&Atile[0][0],
                       (unsigned)(size_t)(void*)&Atile[1][0]};
  u32x8 g1 = g1_a_tile();

  const v8f vz = {0.f,0.f,0.f,0.f,0.f,0.f,0.f,0.f};
  v8f acc[2][4] = {{vz, vz, vz, vz}, {vz, vz, vz, vz}};

  if (wave == 0)
    tdm_load_2d(lds_a[0], (unsigned long long)(size_t)&A[(size_t)m0 * K], g1);

  for (int kt = 0; kt < KT; kt++) {
    int k0 = kt * 32;
    if (wave == 0) {
      if (kt + 1 < KT) {
        tdm_load_2d(lds_a[(kt + 1) & 1],
                    (unsigned long long)(size_t)&A[(size_t)m0 * K + (k0 + 32)], g1);
        __builtin_amdgcn_s_wait_tensorcnt(1);
      } else {
        __builtin_amdgcn_s_wait_tensorcnt(0);
      }
    }
    if (kt + 1 < KT)
      __builtin_prefetch(&Wt[(size_t)(n0 + (lane & 15)) * K + k0 + 32], 0, 1);
    __syncthreads();
    const unsigned short* at = &Atile[kt & 1][0];
    v16bf a0 = load_afrag(at + (waveM * 32) * 32, 32, lane);
    v16bf a1 = load_afrag(at + (waveM * 32 + 16) * 32, 32, lane);
#pragma unroll
    for (int j = 0; j < 4; j++) {
      v16bf b = load_bfrag(&Wt[(size_t)(n0 + j * 16) * K + k0], K, lane);
      acc[0][j] = WMMA_BF16(a0, b, acc[0][j]);
      acc[1][j] = WMMA_BF16(a1, b, acc[1][j]);
    }
    __syncthreads();
  }

  int n = lane & 15, half = lane >> 4;
#pragma unroll
  for (int j = 0; j < 4; j++) {
    int c = n0 + j * 16 + n;
    float bv = bias[c];
#pragma unroll
    for (int s = 0; s < 2; s++) {
#pragma unroll
      for (int r = 0; r < 8; r++) {
        int row = m0 + waveM * 32 + s * 16 + half * 8 + r;
        out[(size_t)row * CCH + c] = acc[s][j][r] + bv;
      }
    }
  }
}

// ---------------- flash attention: one wave per 16-query tile ----------------
__global__ __launch_bounds__(32) void flash_attn_wmma(
    const unsigned short* __restrict__ Qd, const unsigned short* __restrict__ Kd,
    const unsigned short* __restrict__ Vd, unsigned short* __restrict__ Y) {
  int lane = threadIdx.x;
  int q0 = blockIdx.x * 16;
  int bh = blockIdx.y;                  // b*H + h
  int b = bh >> 4, h = bh & 15;
  const unsigned short* qh = Qd + (size_t)bh * TT * HHD;
  const unsigned short* kh = Kd + (size_t)bh * TT * HHD;
  const unsigned short* vh = Vd + (size_t)bh * TT * HHD;

  __shared__ unsigned short Vlds[2][32 * HHD];  // double-buffered V tiles (2x4KB)
  __shared__ unsigned short Plds[16 * 32];      // 16 queries x 32 keys (probs)
  unsigned lds_v[2] = {(unsigned)(size_t)(void*)&Vlds[0][0],
                       (unsigned)(size_t)(void*)&Vlds[1][0]};

  // D# group1: 1D contiguous 4KB tile (2048 x 2B elements)
  u32x8 g1v;
  g1v[0] = 0x00010000u;   // data_size = 2B
  g1v[1] = 0xFFFF0000u;   // tensor_dim0 = 65535 (ample; tile always in-bounds)
  g1v[2] = 0u;
  g1v[3] = 0x08000000u;   // tile_dim0 = 2048
  g1v[4] = 0u;            // tile_dim1 = 0 -> 1D
  g1v[5] = 0xFFFFu;       // tensor_dim0_stride (unused for 1D)
  g1v[6] = 0u;
  g1v[7] = 0u;

  v16bf qf0 = load_afrag(qh + (size_t)q0 * HHD + 0, HHD, lane);
  v16bf qf1 = load_afrag(qh + (size_t)q0 * HHD + 32, HHD, lane);

  const v8f vz = {0.f,0.f,0.f,0.f,0.f,0.f,0.f,0.f};
  v8f o[4] = {vz, vz, vz, vz};
  float mi[8], li[8];
#pragma unroll
  for (int r = 0; r < 8; r++) { mi[r] = -1e30f; li[r] = 0.f; }

  int n = lane & 15, half = lane >> 4, rowbase = half * 8;
  const float scale = 0.125f;   // 1/sqrt(64)

  const int nsteps = (q0 + 16 + 31) >> 5;   // 32-key steps up to the causal edge
  tdm_load_2d(lds_v[0], (unsigned long long)(size_t)vh, g1v);   // V tile 0 in flight

  for (int it = 0; it < nsteps; it++) {
    int kb = it * 32;

    // S = Q Kt over 32 keys as two 16x16 tiles, K(dim)-chunks of 32
    v8f s0 = vz, s1 = vz;
    s0 = WMMA_BF16(qf0, load_bfrag(kh + (size_t)kb * HHD + 0,  HHD, lane), s0);
    s0 = WMMA_BF16(qf1, load_bfrag(kh + (size_t)kb * HHD + 32, HHD, lane), s0);
    s1 = WMMA_BF16(qf0, load_bfrag(kh + (size_t)(kb + 16) * HHD + 0,  HHD, lane), s1);
    s1 = WMMA_BF16(qf1, load_bfrag(kh + (size_t)(kb + 16) * HHD + 32, HHD, lane), s1);

    bool needmask = (kb + 31 > q0);
#pragma unroll
    for (int r = 0; r < 8; r++) {
      float v0 = s0[r] * scale, v1 = s1[r] * scale;
      if (needmask) {
        int query = q0 + rowbase + r;
        if (kb + n      > query) v0 = -1e30f;
        if (kb + 16 + n > query) v1 = -1e30f;
      }
      // row max across the 16 lanes of this row group (wave32: xor 1..8)
      float mx = fmaxf(v0, v1);
      mx = fmaxf(mx, __shfl_xor(mx, 1, 32));
      mx = fmaxf(mx, __shfl_xor(mx, 2, 32));
      mx = fmaxf(mx, __shfl_xor(mx, 4, 32));
      mx = fmaxf(mx, __shfl_xor(mx, 8, 32));
      float mn = fmaxf(mi[r], mx);
      float alpha = __expf(mi[r] - mn);
      float p0 = __expf(v0 - mn);
      float p1 = __expf(v1 - mn);
      float rs = p0 + p1;
      rs += __shfl_xor(rs, 1, 32);
      rs += __shfl_xor(rs, 2, 32);
      rs += __shfl_xor(rs, 4, 32);
      rs += __shfl_xor(rs, 8, 32);
      li[r] = li[r] * alpha + rs;
      mi[r] = mn;
#pragma unroll
      for (int j = 0; j < 4; j++) o[j][r] *= alpha;
      int mrow = rowbase + r;
      Plds[mrow * 32 + n]      = f2bf(p0);
      Plds[mrow * 32 + 16 + n] = f2bf(p1);
    }

    // pipeline: launch V tile it+1, then wait so tile `it` is resident while
    // tile it+1 overlaps the PV WMMAs and the next score/softmax phase
    if (it + 1 < nsteps) {
      tdm_load_2d(lds_v[(it + 1) & 1],
                  (unsigned long long)(size_t)(vh + (size_t)(kb + 32) * HHD), g1v);
      __builtin_amdgcn_s_wait_tensorcnt(1);
    } else {
      __builtin_amdgcn_s_wait_tensorcnt(0);
    }
    __syncthreads();

    // O += P @ V : A = P (16x32), B = V chunks (32x16), 4 dim-chunks
    const unsigned short* vt = &Vlds[it & 1][0];
    v16bf pf = load_afrag(&Plds[0], 32, lane);
#pragma unroll
    for (int j = 0; j < 4; j++) {
      AB16 u;
#pragma unroll
      for (int i = 0; i < 16; i++)
        u.h[i] = vt[(half * 16 + i) * HHD + j * 16 + n];
      o[j] = WMMA_BF16(pf, u.v, o[j]);
    }
    __syncthreads();
  }

  // normalize and write y[b, q0+m, h*64 + dim] as bf16
#pragma unroll
  for (int r = 0; r < 8; r++) {
    int mrow = rowbase + r;
    float inv = 1.0f / li[r];
    size_t rowoff = ((size_t)b * TT + q0 + mrow) * CCH + h * HHD;
#pragma unroll
    for (int j = 0; j < 4; j++)
      Y[rowoff + j * 16 + n] = f2bf(o[j][r] * inv);
  }
}

// ---------------- host launcher ----------------
extern "C" void kernel_launch(void* const* d_in, const int* in_sizes, int n_in,
                              void* d_out, int out_size, void* d_ws, size_t ws_size,
                              hipStream_t stream) {
  (void)in_sizes; (void)n_in; (void)out_size; (void)ws_size;
  const float* x      = (const float*)d_in[0];
  const float* w_attn = (const float*)d_in[1];
  const float* b_attn = (const float*)d_in[2];
  const float* w_proj = (const float*)d_in[3];
  const float* b_proj = (const float*)d_in[4];
  float* out = (float*)d_out;

  // workspace carve (~92 MB total), all bf16-as-ushort
  char* ws = (char*)d_ws;
  size_t off = 0;
  auto carve = [&](size_t bytes) {
    void* p = ws + off;
    off += (bytes + 255) & ~(size_t)255;
    return p;
  };
  unsigned short* xb  = (unsigned short*)carve((size_t)MROWS * CCH * 2);  // x bf16
  unsigned short* wat = (unsigned short*)carve((size_t)NQKV * CCH * 2);   // w_attn^T bf16
  unsigned short* wpt = (unsigned short*)carve((size_t)CCH * CCH * 2);    // w_proj^T bf16
  unsigned short* Qb  = (unsigned short*)carve((size_t)MROWS * CCH * 2);  // [B,H,T,HD]
  unsigned short* Kb  = (unsigned short*)carve((size_t)MROWS * CCH * 2);
  unsigned short* Vb  = (unsigned short*)carve((size_t)MROWS * CCH * 2);
  unsigned short* Yb  = (unsigned short*)carve((size_t)MROWS * CCH * 2);  // attn out bf16

  const int nx = MROWS * CCH;                                  // 8,388,608
  cvt_f32_bf16<<<nx / (256 * 4), 256, 0, stream>>>(x, xb, nx);
  cvt_transpose_bf16<<<(NQKV * CCH) / 256, 256, 0, stream>>>(w_attn, wat, CCH, NQKV);
  cvt_transpose_bf16<<<(CCH * CCH) / 256, 256, 0, stream>>>(w_proj, wpt, CCH, CCH);

  gemm_qkv_wmma<<<dim3(MROWS / 64, NQKV / 256), 256, 0, stream>>>(xb, wat, b_attn, Qb, Kb, Vb);

  flash_attn_wmma<<<dim3(TT / 16, BB * NHEAD), 32, 0, stream>>>(Qb, Kb, Vb, Yb);

  gemm_proj_wmma<<<dim3(MROWS / 64, CCH / 256), 256, 0, stream>>>(Yb, wpt, b_proj, out);
}
